// GRU_37761352466675
// MI455X (gfx1250) — matrix operand order repaired and checked by reading, not compiled, not run on hardware
//
#include <hip/hip_runtime.h>
#include <hip/hip_bf16.h>

typedef __attribute__((ext_vector_type(16))) _Float16     v16h;
typedef __attribute__((ext_vector_type(8)))  float        v8f;
typedef __attribute__((ext_vector_type(4)))  unsigned int u32x4;
typedef __attribute__((ext_vector_type(8)))  int          i32x8;
typedef __attribute__((ext_vector_type(4)))  int          i32x4;

#define B_DIM   2
#define S_DIM   2048
#define DIN     2048
#define STATE   2048
#define H_HEADS 16
#define D_HEAD  128
#define C3      1536      // 3 * HG * D
#define MTOT    4096      // B * S
#define FACTOR  0.5f      // 8/sqrt(2*128)

// ---------------------------------------------------------------------------
// TDM: async 2D tile load (rows x cols f16, row stride in halves) -> LDS.
// Builds the D# per CDNA5 ISA 8.3/8.4: group0 {count=1, lds_addr, global_addr,
// type=2}, group1 {data_size=2B, dims, tile dims, dim0 stride}, groups 2/3 = 0
// (2D tile). One instruction per wave; tracked by TENSORcnt.
// ---------------------------------------------------------------------------
__device__ __forceinline__ void tdm_load_2d_f16(unsigned lds_off, const _Float16* gptr,
                                                unsigned rows, unsigned cols,
                                                unsigned stride_halves)
{
    const unsigned long long ga = (unsigned long long)(size_t)gptr;

    u32x4 g0;
    g0[0] = 1u;                                                  // count=1, user mode
    g0[1] = lds_off;                                             // lds_addr (bytes)
    g0[2] = (unsigned)(ga & 0xFFFFFFFFu);                        // global_addr[31:0]
    g0[3] = (unsigned)((ga >> 32) & 0x01FFFFFFu) | (2u << 30);   // addr[56:32] | type=2

    const unsigned td0 = 1u << 20, td1 = 1u << 20;               // generous tensor bounds
    i32x8 g1;
    g1[0] = (int)(1u << 16);                                     // data_size = 2 bytes
    g1[1] = (int)((td0 & 0xFFFFu) << 16);                        // tensor_dim0 lo16
    g1[2] = (int)((td0 >> 16) | ((td1 & 0xFFFFu) << 16));        // dim0 hi / dim1 lo
    g1[3] = (int)((td1 >> 16) | (cols << 16));                   // dim1 hi | tile_dim0
    g1[4] = (int)rows;                                           // tile_dim1 (tile_dim2=0)
    g1[5] = (int)stride_halves;                                  // tensor_dim0_stride[31:0]
    g1[6] = 0;                                                   // stride hi | dim1_stride lo
    g1[7] = 0;

    const i32x4 z4 = {0, 0, 0, 0};
#if __clang_major__ >= 23
    const i32x8 z8 = {0, 0, 0, 0, 0, 0, 0, 0};
    __builtin_amdgcn_tensor_load_to_lds(g0, g1, z4, z4, z8, 0);
#else
    __builtin_amdgcn_tensor_load_to_lds(g0, g1, z4, z4, 0);
#endif
}

// ---------------------------------------------------------------------------
// f32 -> f16 conversion (weights & activations for WMMA)
// ---------------------------------------------------------------------------
__global__ void f32_to_f16(const float* __restrict__ in, _Float16* __restrict__ out, long n) {
    long i  = (long)blockIdx.x * blockDim.x + threadIdx.x;
    long st = (long)gridDim.x * blockDim.x;
    for (; i < n; i += st) out[i] = (_Float16)in[i];
}

// ---------------------------------------------------------------------------
// Generic tiled WMMA GEMM:  C[g] = A[g] (MxK f16) * B[g] (KxN f16) [*scale +bias]
// Tile: BM=128, BN=128, BK=32. 256 threads = 8 waves, each wave 32x64 (2x4 frags).
// A tile staged by TDM (each wave DMAs its own 16-row slice), B tile staged
// transposed [n][k] by vector loads + LDS scatter.
// ---------------------------------------------------------------------------
template<bool OUT_F16, bool HAS_BIAS>
__launch_bounds__(256)
__global__ void wmma_gemm(const _Float16* __restrict__ A, long strideAg, int lda,
                          const _Float16* __restrict__ Bm, long strideBg, int ldb,
                          void* __restrict__ C, long strideCg, int ldc,
                          const float* __restrict__ bias,
                          int Kdim, float scale)
{
    __shared__ _Float16 sA[128][32];
    __shared__ _Float16 sB[128][34];   // [n][k], padded to dodge bank conflicts

    const int tid  = threadIdx.x;
    const int wave = tid >> 5;
    const int lane = tid & 31;
    const int wm   = wave >> 1;        // 0..3 : 32-row strip
    const int wn   = wave & 1;         // 0..1 : 64-col strip
    const int blockM = blockIdx.y * 128;
    const int blockN = blockIdx.x * 128;
    const int g      = blockIdx.z;

    const _Float16* Ag = A  + (long)g * strideAg;
    const _Float16* Bg = Bm + (long)g * strideBg;

    v8f acc[2][4];
    #pragma unroll
    for (int mt = 0; mt < 2; ++mt)
        #pragma unroll
        for (int nt = 0; nt < 4; ++nt)
            #pragma unroll
            for (int i = 0; i < 8; ++i) acc[mt][nt][i] = 0.0f;

    const int bkrow = tid >> 3, bcol = tid & 7;    // B: 32 rows x 8 col-chunks of 16 halves

    const int am  = lane & 15;
    const int kba = (lane < 16) ? 0 : 16;          // A frag K-base (contiguous K order, matches B)
    const int kbb = (lane < 16) ? 0 : 16;          // B frag K-base (contiguous K order, matches A)

    // per-wave TDM destination: 16-row slice of the A tile
    const unsigned ldsA_off = (unsigned)(size_t)(&sA[wave * 16][0]);
    const _Float16* Arow    = Ag + (long)(blockM + wave * 16) * lda;

    for (int k0 = 0; k0 < Kdim; k0 += 32) {
        // ---- stage A tile: one TDM descriptor per wave (16 x 32 f16 slice) ----
        tdm_load_2d_f16(ldsA_off, Arow + k0, 16, 32, (unsigned)lda);

        // ---- stage B tile transposed into sB[n][k] ----
        v16h tb = *reinterpret_cast<const v16h*>(Bg + (long)(k0 + bkrow) * ldb + blockN + bcol * 16);
        #pragma unroll
        for (int j = 0; j < 16; ++j) sB[bcol * 16 + j][bkrow] = tb[j];

        if (k0 + 32 < Kdim)   // hint next B tile toward L2 (global_prefetch_b8)
            __builtin_prefetch(Bg + (long)(k0 + 32 + bkrow) * ldb + blockN + bcol * 16, 0, 1);

        __builtin_amdgcn_s_wait_tensorcnt(0);   // our wave's A slice is in LDS
        __syncthreads();                        // everyone's A+B staged

        // ---- load fragments ----
        v16h afr[2], bfr[4];
        #pragma unroll
        for (int mt = 0; mt < 2; ++mt) {
            const int row = wm * 32 + mt * 16 + am;
            #pragma unroll
            for (int j = 0; j < 16; ++j) {
                afr[mt][j] = sA[row][kba + j];
            }
        }
        #pragma unroll
        for (int nt = 0; nt < 4; ++nt) {
            const int col = wn * 64 + nt * 16 + am;
            #pragma unroll
            for (int j = 0; j < 16; ++j) bfr[nt][j] = sB[col][kbb + j];
        }

        // ---- 8 WMMAs per wave per k-step ----
        #pragma unroll
        for (int mt = 0; mt < 2; ++mt)
            #pragma unroll
            for (int nt = 0; nt < 4; ++nt)
                acc[mt][nt] = __builtin_amdgcn_wmma_f32_16x16x32_f16(
                    false, afr[mt], false, bfr[nt], (short)0, acc[mt][nt], false, false);

        __syncthreads();
    }

    // ---- store C (ISA C/D layout: vgpr i -> row i or i+8; col = lane&15) ----
    const int cn = lane & 15;
    #pragma unroll
    for (int mt = 0; mt < 2; ++mt) {
        #pragma unroll
        for (int nt = 0; nt < 4; ++nt) {
            #pragma unroll
            for (int i = 0; i < 8; ++i) {
                const int rl  = wm * 32 + mt * 16 + ((lane < 16) ? i : i + 8);
                const int cl  = wn * 64 + nt * 16 + cn;
                const long row = blockM + rl;
                const int  col = blockN + cl;
                float v = acc[mt][nt][i] * scale;
                if (HAS_BIAS) v += bias[col];
                const long idx = (long)g * strideCg + row * (long)ldc + col;
                if (OUT_F16) ((_Float16*)C)[idx] = (_Float16)v;
                else         ((float*)C)[idx]    = v;
            }
        }
    }
}

// ---------------------------------------------------------------------------
// Depthwise causal conv (K=4) along S + RMSNorm over 1536 channels -> f16
// one block per (b,s); 256 threads x 6 channels
// ---------------------------------------------------------------------------
__global__ void conv_rms(const float* __restrict__ h0, const float* __restrict__ cw,
                         const float* __restrict__ rw, _Float16* __restrict__ h1)
{
    const int m = blockIdx.x;
    const int b = m / S_DIM, s = m % S_DIM;
    const int tid = threadIdx.x;
    __shared__ float red[256];

    float v[6];
    float ss = 0.0f;
    #pragma unroll
    for (int i = 0; i < 6; ++i) {
        const int c = tid + i * 256;
        float a = 0.0f;
        #pragma unroll
        for (int j = 0; j < 4; ++j) {
            const int sp = s - 3 + j;
            const float xv = (sp >= 0) ? h0[((long)b * S_DIM + sp) * C3 + c] : 0.0f;
            a += xv * cw[c * 4 + j];
        }
        v[i] = a;
        ss += a * a;
    }
    red[tid] = ss;
    __syncthreads();
    for (int o = 128; o > 0; o >>= 1) {
        if (tid < o) red[tid] += red[tid + o];
        __syncthreads();
    }
    const float inv = rsqrtf(red[0] / (float)C3 + 1e-6f);
    #pragma unroll
    for (int i = 0; i < 6; ++i) {
        const int c = tid + i * 256;
        h1[(long)m * C3 + c] = (_Float16)(v[i] * rw[c] * inv);
    }
}

// ---------------------------------------------------------------------------
// RMSNorm over 2048 channels (f32 in -> f16 out); 256 threads x 8 channels
// ---------------------------------------------------------------------------
__global__ void rms2048(const float* __restrict__ y, const float* __restrict__ rw,
                        _Float16* __restrict__ yn)
{
    const int m = blockIdx.x;
    const int tid = threadIdx.x;
    __shared__ float red[256];

    float v[8];
    float ss = 0.0f;
    #pragma unroll
    for (int i = 0; i < 8; ++i) {
        const int c = tid + i * 256;
        v[i] = y[(long)m * STATE + c];
        ss += v[i] * v[i];
    }
    red[tid] = ss;
    __syncthreads();
    for (int o = 128; o > 0; o >>= 1) {
        if (tid < o) red[tid] += red[tid + o];
        __syncthreads();
    }
    const float inv = rsqrtf(red[0] / (float)STATE + 1e-6f);
    #pragma unroll
    for (int i = 0; i < 8; ++i) {
        const int c = tid + i * 256;
        yn[(long)m * STATE + c] = (_Float16)(v[i] * rw[c] * inv);
    }
}

// ---------------------------------------------------------------------------
// Serial gated recurrence. One block per (b,h); 128 threads = one state elem.
// Wz/Wf/Wr (*FACTOR) resident in 192 KB dynamic LDS (CDNA5: 320 KB/WGP).
// xall holds pre-scaled gate inputs (m, 6144) f16; writes y (m, 2048) f16.
// ---------------------------------------------------------------------------
__global__ void recurrence(const _Float16* __restrict__ xall,
                           const float* __restrict__ sw,
                           const float* __restrict__ fb, const float* __restrict__ rb,
                           _Float16* __restrict__ yrec)
{
    extern __shared__ float sm[];
    float* Wz  = sm;
    float* Wf  = sm + 16384;
    float* Wr  = sm + 32768;
    float* hsh = sm + 49152;   // 128 floats of running state

    const int bh = blockIdx.x;
    const int b  = bh >> 4;
    const int h  = bh & 15;
    const int e  = threadIdx.x;     // 0..127

    for (int i = e; i < 16384; i += 128) {
        Wz[i] = sw[(long)h        * 16384 + i] * FACTOR;
        Wf[i] = sw[(long)(16 + h) * 16384 + i] * FACTOR;
        Wr[i] = sw[(long)(32 + h) * 16384 + i] * FACTOR;
    }
    hsh[e] = 0.0f;
    const float fbias = fb[h * 128 + e];
    const float rbias = rb[h * 128 + e];
    __syncthreads();

    for (int t = 0; t < S_DIM; ++t) {
        const long base = ((long)b * S_DIM + t) * 6144 + h * 128 + e;
        const float xi = (float)xall[base];
        const float xf = (float)xall[base + 2048] + fbias;
        const float xr = (float)xall[base + 4096] + rbias;

        float az = 0.0f, af = 0.0f, ar = 0.0f;
        #pragma unroll 8
        for (int d = 0; d < 128; ++d) {
            const float hd = hsh[d];          // LDS broadcast read
            az += hd * Wz[d * 128 + e];       // lanes hit distinct banks
            af += hd * Wf[d * 128 + e];
            ar += hd * Wr[d * 128 + e];
        }
        const float r  = 1.0f / (1.0f + __expf(-(xr + ar)));
        const float f  = 1.0f / (1.0f + __expf(-(xf + af)));
        const float nn = tanhf(xi + r * az);
        const float hv = f * hsh[e] + (1.0f - f) * nn;

        yrec[((long)b * S_DIM + t) * STATE + h * 128 + e] = (_Float16)hv;
        __syncthreads();
        hsh[e] = hv;
        __syncthreads();
    }
}

// ---------------------------------------------------------------------------
extern "C" void kernel_launch(void* const* d_in, const int* in_sizes, int n_in,
                              void* d_out, int out_size, void* d_ws, size_t ws_size,
                              hipStream_t stream)
{
    (void)in_sizes; (void)n_in; (void)out_size; (void)ws_size;

    const float* x        = (const float*)d_in[0];
    const float* W_in     = (const float*)d_in[1];
    const float* b_in     = (const float*)d_in[2];
    const float* conv_w   = (const float*)d_in[3];
    const float* rms1_w   = (const float*)d_in[4];
    const float* head_w   = (const float*)d_in[5];
    const float* state_w  = (const float*)d_in[6];
    const float* fbias    = (const float*)d_in[7];
    const float* rbias    = (const float*)d_in[8];
    const float* out_hw   = (const float*)d_in[9];
    const float* rms2_w   = (const float*)d_in[10];
    const float* W_out    = (const float*)d_in[11];
    float* out = (float*)d_out;

    char* ws = (char*)d_ws;
    size_t off = 0;
    auto carve = [&](size_t bytes) -> char* {
        char* p = ws + off;
        off += (bytes + 255) & ~(size_t)255;
        return p;
    };

    const long nX   = (long)MTOT * DIN;        // 8388608
    const long nWin = (long)DIN * C3;          // 3145728
    const long nHW  = 12L * 128 * 512;         // 786432
    const long nOHW = 16L * 128 * 128;         // 262144
    const long nWo  = (long)STATE * STATE;     // 4194304

    _Float16* xf16    = (_Float16*)carve(nX * 2);
    _Float16* Winf16  = (_Float16*)carve(nWin * 2);
    _Float16* hwf16   = (_Float16*)carve(nHW * 2);
    _Float16* ohwf16  = (_Float16*)carve(nOHW * 2);
    _Float16* Wof16   = (_Float16*)carve(nWo * 2);
    float*    h0      = (float*)   carve((long)MTOT * C3 * 4);
    _Float16* h1f16   = (_Float16*)carve((long)MTOT * C3 * 2);
    _Float16* h6144   = (_Float16*)carve((long)MTOT * 6144 * 2);
    _Float16* yrec    = (_Float16*)carve((long)MTOT * STATE * 2);
    float*    y2      = (float*)   carve((long)MTOT * STATE * 4);
    _Float16* y2n     = (_Float16*)carve((long)MTOT * STATE * 2);

    // 0) f16 conversions
    f32_to_f16<<<2048, 256, 0, stream>>>(x,      xf16,   nX);
    f32_to_f16<<<2048, 256, 0, stream>>>(W_in,   Winf16, nWin);
    f32_to_f16<<<512,  256, 0, stream>>>(head_w, hwf16,  nHW);
    f32_to_f16<<<512,  256, 0, stream>>>(out_hw, ohwf16, nOHW);
    f32_to_f16<<<2048, 256, 0, stream>>>(W_out,  Wof16,  nWo);

    // 1) h0 = x @ W_in + b_in              (4096x2048)x(2048x1536) -> f32
    wmma_gemm<false, true><<<dim3(12, 32, 1), 256, 0, stream>>>(
        xf16, 0, DIN, Winf16, 0, C3, h0, 0, C3, b_in, DIN, 1.0f);

    // 2) causal depthwise conv + rmsnorm -> h1 f16
    conv_rms<<<MTOT, 256, 0, stream>>>(h0, conv_w, rms1_w, h1f16);

    // 3) grouped head projection (12 groups of (4096x128)x(128x512)), *FACTOR -> h6144 f16
    wmma_gemm<true, false><<<dim3(4, 32, 12), 256, 0, stream>>>(
        h1f16, 128, C3, hwf16, 128L * 512, 512, h6144, 512, 6144, nullptr, 128, FACTOR);

    // 4) serial recurrence: 32 blocks (b,h), 192 KB LDS weights + state
    recurrence<<<32, 128, (3 * 16384 + 128) * sizeof(float), stream>>>(
        h6144, state_w, fbias, rbias, yrec);

    // 5) output head projection (16 groups of (4096x128)x(128x128)) -> y2 f32
    wmma_gemm<false, false><<<dim3(1, 32, 16), 256, 0, stream>>>(
        yrec, 128, STATE, ohwf16, 128L * 128, 128, y2, 128, STATE, nullptr, 128, 1.0f);

    // 6) rmsnorm over 2048 -> y2n f16
    rms2048<<<MTOT, 256, 0, stream>>>(y2, rms2_w, y2n);

    // 7) final GEMM: (4096x2048)x(2048x2048) -> out f32
    wmma_gemm<false, false><<<dim3(16, 32, 1), 256, 0, stream>>>(
        y2n, 0, STATE, Wof16, 0, STATE, out, 0, STATE, nullptr, STATE, 1.0f);
}
